// TTGNN_40243843563861
// MI455X (gfx1250) — matrix operand order repaired and checked by reading, not compiled, not run on hardware
//
#include <hip/hip_runtime.h>
#include <hip/hip_bf16.h>

// ---------------- problem constants ----------------
static constexpr int kN = 10000;
static constexpr int kE = 80000;
static constexpr int kD = 1024;
static constexpr int kH = 8;
static constexpr int kC = 128;
static constexpr int kL = 2;
static constexpr int kNE = kE + kN;   // edges + self loops
static constexpr float kNEG = 0.2f;

// ---------------- WMMA vector types ----------------
typedef __attribute__((ext_vector_type(16))) __bf16 v16bf;
typedef __attribute__((ext_vector_type(8)))  float  v8f;

// fp32 -> bf16 (round to nearest even), packed pair (lo in [15:0], hi in [31:16])
__device__ __forceinline__ unsigned pack_bf16(float lo, float hi) {
  unsigned ul = __float_as_uint(lo);
  ul += 0x7FFFu + ((ul >> 16) & 1u);
  unsigned uh = __float_as_uint(hi);
  uh += 0x7FFFu + ((uh >> 16) & 1u);
  return (ul >> 16) | (uh & 0xFFFF0000u);
}

// monotone float <-> uint key for atomicMax-based segment max
__device__ __forceinline__ unsigned fkey(float f) {
  unsigned u = __float_as_uint(f);
  return (u & 0x80000000u) ? ~u : (u | 0x80000000u);
}
__device__ __forceinline__ float kinv(unsigned k) {
  unsigned u = (k & 0x80000000u) ? (k & 0x7FFFFFFFu) : ~k;
  return __uint_as_float(u);
}

// ---------------- utility fills ----------------
__global__ void fill_f32(float* p, float v, int n) {
  int i = blockIdx.x * blockDim.x + threadIdx.x;
  if (i < n) p[i] = v;
}
__global__ void fill_u32(unsigned* p, unsigned v, int n) {
  int i = blockIdx.x * blockDim.x + threadIdx.x;
  if (i < n) p[i] = v;
}

// ---------------- h = x + node_type_emb[node_types] ----------------
__global__ void embed_nodes(const float* __restrict__ x, const int* __restrict__ ntyp,
                            const float* __restrict__ ntemb, float* __restrict__ h, int total) {
  int i = blockIdx.x * blockDim.x + threadIdx.x;
  if (i >= total) return;
  int n = i >> 10, j = i & 1023;
  h[i] = x[i] + ntemb[ntyp[n] * kD + j];
}

// ---------------- per-node incoming edge-type counts ----------------
__global__ void count_types(const int* __restrict__ ei, const int* __restrict__ eattr,
                            int* __restrict__ typecnt, int E) {
  int e = blockIdx.x * blockDim.x + threadIdx.x;
  if (e >= E) return;
  atomicAdd(&typecnt[ei[E + e] * 3 + eattr[e]], 1);
}

// ---------------- Wee = edge_type_emb @ We  (3 x 1024) ----------------
__global__ void edge_weight_rows(const float* __restrict__ etemb, const float* __restrict__ WeL,
                                 float* __restrict__ wee) {
  int i = blockIdx.x * blockDim.x + threadIdx.x;
  if (i >= 3 * kD) return;
  int t = i >> 10, j = i & 1023;
  const float* er = etemb + t * kD;
  float s = 0.f;
  for (int k = 0; k < kD; ++k) s = fmaf(er[k], WeL[k * kD + j], s);
  wee[i] = s;
}

// ---------------- bf16 WMMA GEMM: out[M x 1024] = A[M x 1024] @ W[1024 x 1024] + bias ----
// Workgroup: 256 threads = 8 wave32s; block tile 64(M) x 128(N); K stepped by 64.
// 8 independent v_wmma_f32_16x16x32_bf16 per barrier pair; LDS holds A/B tiles as
// packed bf16 K-pairs matching the CDNA5 16-bit WMMA operand layouts.
// A rows are CLAMPED (not exec-masked) at the M edge: pad-row garbage only feeds
// accumulator rows whose stores are guarded, so the hot loop stays branch-free.
__global__ __launch_bounds__(256)
void gemm_bf16_wmma(const float* __restrict__ A, const float* __restrict__ W,
                    const float* __restrict__ bias, float* __restrict__ out, int M) {
  __shared__ unsigned ldsA[64 * 32];    // [row][kpair 0..31]   (8 KB)
  __shared__ unsigned ldsB[128 * 32];   // [col][kpair 0..31]   (16 KB)

  const int tid   = threadIdx.x;
  const int lane  = tid & 31;
  const int wid   = tid >> 5;
  const int waveM = wid & 3;            // 4 waves along M
  const int waveN = wid >> 2;           // 2 waves along N (each covers 64 cols)
  const int bm = blockIdx.x * 64;
  const int bn = blockIdx.y * 128;

  v8f acc[4];
  const v8f vzero = {0.f, 0.f, 0.f, 0.f, 0.f, 0.f, 0.f, 0.f};
#pragma unroll
  for (int a = 0; a < 4; ++a) acc[a] = vzero;

  // per-thread staging coordinates (constant across the K loop)
  const int aRow = tid >> 5;            // A: 8 rows per pass, 32 pairs wide
  const int aT   = tid & 31;
  const int aRowG = min(bm + aRow, M - 1);      // clamp instead of branch
  const int bT   = tid >> 7;            // B: 2 k-pairs per pass, 128 cols wide
  const int bCol = tid & 127;
  const int bColG = bn + bCol;

  for (int k0 = 0; k0 < kD; k0 += 64) {
    // ---- stage A tile: 64 rows x 32 pairs (fp32 -> bf16 pack), branch-free ----
#pragma unroll
    for (int r = 0; r < 8; ++r) {
      int row = min(aRowG + r * 8, M - 1);
      int gk = k0 + 2 * aT;
      float lo = A[(size_t)row * kD + gk];
      float hi = A[(size_t)row * kD + gk + 1];
      ldsA[(aRow + r * 8) * 32 + aT] = pack_bf16(lo, hi);
    }
    // ---- stage B tile: 64 K x 128 cols, stored [col][kpair] for b128 frag reads ----
#pragma unroll
    for (int r = 0; r < 16; ++r) {
      int t = bT + r * 2;               // k-pair 0..31 (coalesced across cols)
      int gk = k0 + 2 * t;
      float lo = W[(size_t)gk * kD + bColG];
      float hi = W[(size_t)(gk + 1) * kD + bColG];
      ldsB[bCol * 32 + t] = pack_bf16(lo, hi);
    }
    if (k0 + 64 < kD)
      __builtin_prefetch(&W[(size_t)(k0 + 64 + 2 * bT) * kD + bColG], 0, 0);
    __syncthreads();

    const int rowb = (waveM * 16 + (lane & 15)) * 32;
    const int g4 = (lane & 16) ? 4 : 0;   // A: lanes 0-15 K∈{0-7,16-23}, 16-31 K∈{8-15,24-31}
    const int tB = (lane & 16) ? 4 : 0;   // B: lanes 0-15 K∈{0-7,16-23}, 16-31 K∈{8-15,24-31}
#pragma unroll
    for (int q = 0; q < 2; ++q) {         // two K=32 sub-steps per staged tile
      union { unsigned u[8]; v16bf v; } fa;
      *(uint4*)&fa.u[0] = *(const uint4*)&ldsA[rowb + q * 16 + g4];
      *(uint4*)&fa.u[4] = *(const uint4*)&ldsA[rowb + q * 16 + 8 + g4];
#pragma unroll
      for (int a = 0; a < 4; ++a) {
        union { unsigned u[8]; v16bf v; } fb;
        int colb = (waveN * 64 + a * 16 + (lane & 15)) * 32 + q * 16;
        *(uint4*)&fb.u[0] = *(const uint4*)&ldsB[colb + tB];
        *(uint4*)&fb.u[4] = *(const uint4*)&ldsB[colb + tB + 8];
        acc[a] = __builtin_amdgcn_wmma_f32_16x16x32_bf16(
            false, fa.v, false, fb.v, (short)0, acc[a], false, false);
      }
    }
    __syncthreads();
  }

  // ---- store: C/D layout — VGPR r: lanes 0-15 M=r, lanes 16-31 M=8+r; N = lane&15 ----
  const int rowBase = bm + waveM * 16 + ((lane & 16) ? 8 : 0);
  const int colL = lane & 15;
#pragma unroll
  for (int a = 0; a < 4; ++a) {
    int gc = bn + waveN * 64 + a * 16 + colL;
    float bv = bias[gc];
#pragma unroll
    for (int r = 0; r < 8; ++r) {
      int gm = rowBase + r;
      if (gm < M) out[(size_t)gm * kD + gc] = acc[a][r] + bv;
    }
  }
}

// ---------------- attention logits + segment max (one wave per (edge,head)) ----------
__global__ __launch_bounds__(256)
void gat_logits(const int* __restrict__ ei, const int* __restrict__ eattr,
                const int* __restrict__ typecnt,
                const float* __restrict__ xl, const float* __restrict__ xr,
                const float* __restrict__ wee, const float* __restrict__ attL,
                float* __restrict__ logit, unsigned* __restrict__ mkey) {
  int gw = blockIdx.x * 8 + (threadIdx.x >> 5);
  int lane = threadIdx.x & 31;
  if (gw >= kNE * kH) return;
  int idx = gw >> 3, head = gw & 7;

  int s, d;
  float w0 = 0.f, w1 = 0.f, w2 = 0.f;
  const float* ep = nullptr;
  bool isEdge = idx < kE;
  if (isEdge) {
    s = ei[idx];
    d = ei[kE + idx];
    ep = wee + eattr[idx] * kD;
  } else {
    int n = idx - kE;
    s = n; d = n;
    float c0 = (float)typecnt[n * 3 + 0];
    float c1 = (float)typecnt[n * 3 + 1];
    float c2 = (float)typecnt[n * 3 + 2];
    float inv = 1.f / fmaxf(c0 + c1 + c2, 1.f);
    w0 = c0 * inv; w1 = c1 * inv; w2 = c2 * inv;
  }

  const float* xls = xl + (size_t)s * kD + head * kC;
  const float* xrd = xr + (size_t)d * kD + head * kC;
  const float* av  = attL + head * kC;
  float acc = 0.f;
#pragma unroll
  for (int j = 0; j < 4; ++j) {
    int c = lane + j * 32;
    float epv = isEdge ? ep[head * kC + c]
                       : (w0 * wee[0 * kD + head * kC + c] +
                          w1 * wee[1 * kD + head * kC + c] +
                          w2 * wee[2 * kD + head * kC + c]);
    float v = xls[c] + xrd[c] + epv;
    v = (v > 0.f) ? v : kNEG * v;
    acc = fmaf(av[c], v, acc);
  }
  for (int o = 16; o; o >>= 1) acc += __shfl_xor(acc, o, 32);
  if (lane == 0) {
    logit[gw] = acc;
    atomicMax(&mkey[d * kH + head], fkey(acc));
  }
}

// ---------------- exp / denom / un-normalized aggregation ----------------
__global__ __launch_bounds__(256)
void gat_scatter(const int* __restrict__ ei, const float* __restrict__ xl,
                 const float* __restrict__ logit, const unsigned* __restrict__ mkey,
                 float* __restrict__ denom, float* __restrict__ agg) {
  int gw = blockIdx.x * 8 + (threadIdx.x >> 5);
  int lane = threadIdx.x & 31;
  if (gw >= kNE * kH) return;
  int idx = gw >> 3, head = gw & 7;
  int s, d;
  if (idx < kE) { s = ei[idx]; d = ei[kE + idx]; }
  else          { s = d = idx - kE; }
  float m = kinv(mkey[d * kH + head]);
  float a = __expf(logit[gw] - m);
  if (lane == 0) atomicAdd(&denom[d * kH + head], a);
  const float* xls = xl + (size_t)s * kD + head * kC;
  float* ag = agg + (size_t)d * kD + head * kC;
#pragma unroll
  for (int j = 0; j < 4; ++j) {
    int c = lane + j * 32;
    atomicAdd(&ag[c], a * xls[c]);
  }
}

// ---------------- h += relu(agg/denom + conv_bias) ----------------
__global__ void gat_update(float* __restrict__ h, const float* __restrict__ agg,
                           const float* __restrict__ denom, const float* __restrict__ cb,
                           int total) {
  int i = blockIdx.x * blockDim.x + threadIdx.x;
  if (i >= total) return;
  int n = i >> 10, j = i & 1023, head = j >> 7;
  float o = agg[i] / denom[n * kH + head] + cb[j];
  h[i] += fmaxf(o, 0.f);
}

// ---------------- driver ----------------
extern "C" void kernel_launch(void* const* d_in, const int* in_sizes, int n_in,
                              void* d_out, int out_size, void* d_ws, size_t ws_size,
                              hipStream_t stream) {
  (void)in_sizes; (void)n_in; (void)out_size; (void)ws_size;
  const float* x     = (const float*)d_in[0];
  const int*   ei    = (const int*)d_in[1];
  const int*   eattr = (const int*)d_in[2];
  const int*   ntyp  = (const int*)d_in[3];
  const float* ntemb = (const float*)d_in[4];
  const float* etemb = (const float*)d_in[5];
  const float* Wl    = (const float*)d_in[6];
  const float* bl    = (const float*)d_in[7];
  const float* Wr    = (const float*)d_in[8];
  const float* br    = (const float*)d_in[9];
  const float* We    = (const float*)d_in[10];
  const float* att   = (const float*)d_in[11];
  const float* cbias = (const float*)d_in[12];
  const float* outW  = (const float*)d_in[13];
  const float* outb  = (const float*)d_in[14];
  float* out = (float*)d_out;

  // workspace carve-out
  char* ws = (char*)d_ws;
  size_t off = 0;
  auto carve = [&](size_t bytes) -> char* {
    char* p = ws + off;
    off += (bytes + 255) & ~(size_t)255;
    return p;
  };
  float*    h       = (float*)carve(sizeof(float) * kN * kD);
  float*    xl      = (float*)carve(sizeof(float) * kN * kD);
  float*    xr      = (float*)carve(sizeof(float) * kN * kD);
  float*    agg     = (float*)carve(sizeof(float) * kN * kD);
  float*    logit   = (float*)carve(sizeof(float) * kNE * kH);
  unsigned* mkey    = (unsigned*)carve(sizeof(unsigned) * kN * kH);
  float*    denom   = (float*)carve(sizeof(float) * kN * kH);
  float*    wee     = (float*)carve(sizeof(float) * 3 * kD);
  int*      typecnt = (int*)carve(sizeof(int) * kN * 3);

  auto nb = [](int n, int t) { return (n + t - 1) / t; };
  const dim3 gemmGrid((kN + 63) / 64, kD / 128);
  const int waveBlocks = (kNE * kH + 7) / 8;   // 8 waves per 256-thread block

  // per-node edge-type histogram (fixed across layers)
  fill_u32<<<nb(kN * 3, 256), 256, 0, stream>>>((unsigned*)typecnt, 0u, kN * 3);
  count_types<<<nb(kE, 256), 256, 0, stream>>>(ei, eattr, typecnt, kE);

  // h = x + node_type_emb[node_types]
  embed_nodes<<<nb(kN * kD, 256), 256, 0, stream>>>(x, ntyp, ntemb, h, kN * kD);

  for (int i = 0; i < kL; ++i) {
    // Wee = edge_type_emb @ We[i]  (only 3 distinct edge-feature rows!)
    edge_weight_rows<<<nb(3 * kD, 256), 256, 0, stream>>>(etemb, We + (size_t)i * kD * kD, wee);

    // xl = h @ Wl[i] + bl[i]; xr = h @ Wr[i] + br[i]   (bf16 WMMA, f32 accumulate)
    gemm_bf16_wmma<<<gemmGrid, 256, 0, stream>>>(h, Wl + (size_t)i * kD * kD, bl + i * kD, xl, kN);
    gemm_bf16_wmma<<<gemmGrid, 256, 0, stream>>>(h, Wr + (size_t)i * kD * kD, br + i * kD, xr, kN);

    // reset softmax state + aggregation
    fill_u32<<<nb(kN * kH, 256), 256, 0, stream>>>(mkey, 0u, kN * kH);
    fill_f32<<<nb(kN * kH, 256), 256, 0, stream>>>(denom, 0.f, kN * kH);
    fill_f32<<<nb(kN * kD, 256), 256, 0, stream>>>(agg, 0.f, kN * kD);

    // attention logits + segment max, then exp/denom + weighted aggregation
    gat_logits<<<waveBlocks, 256, 0, stream>>>(ei, eattr, typecnt, xl, xr, wee,
                                               att + (size_t)i * kH * kC, logit, mkey);
    gat_scatter<<<waveBlocks, 256, 0, stream>>>(ei, xl, logit, mkey, denom, agg);

    // h += relu(agg/denom + conv_bias[i])
    gat_update<<<nb(kN * kD, 256), 256, 0, stream>>>(h, agg, denom, cbias + i * kD, kN * kD);
  }

  // out = h @ out_W + out_b
  gemm_bf16_wmma<<<gemmGrid, 256, 0, stream>>>(h, outW, outb, out, kN);
}
